// MultiHeadAttention_51445118271962
// MI455X (gfx1250) — matrix operand order repaired and checked by reading, not compiled, not run on hardware
//
#include <hip/hip_runtime.h>
#include <hip/hip_bf16.h>

typedef __attribute__((ext_vector_type(16))) __bf16 v16bf;
typedef __attribute__((ext_vector_type(8)))  float  v8f;

union FragH {
    uint4 q[2];
    unsigned short u[16];
    v16bf b;
};

__device__ __forceinline__ unsigned short f2bf(float f) {
    union { float f; unsigned int u; } c; c.f = f;
    unsigned int u = c.u;
    unsigned int r = u + 0x7FFFu + ((u >> 16) & 1u);   // round-to-nearest-even
    return (unsigned short)(r >> 16);
}

// ---------------- conversion kernels ----------------

__global__ void cvt_f32_bf16(const float* __restrict__ in,
                             unsigned short* __restrict__ out, int n) {
    int i = blockIdx.x * blockDim.x + threadIdx.x;
    if (i < n) out[i] = f2bf(in[i]);
}

// w is (K x N) row-major; write wT as (N x K) bf16 so WMMA B-fragments are contiguous
__global__ void cvt_transpose_bf16(const float* __restrict__ w,
                                   unsigned short* __restrict__ wT, int K, int N) {
    int i = blockIdx.x * blockDim.x + threadIdx.x;
    if (i < K * N) {
        int k = i / N, n = i % N;
        wT[(size_t)n * K + k] = f2bf(w[i]);
    }
}

// ------- bf16 WMMA GEMM, 32x64 macro-tile per wave (8 WMMAs per K-step) -------
// mode 0: out bf16 [H][S][64] scaled (Q)     mode 1: out bf16 [H][S][64] (K)
// mode 2: out bf16 [H][64][S] transposed (V) mode 3: out f32 [M][N] + bias
__global__ __launch_bounds__(32) void gemm_bf16_wmma(
    const unsigned short* __restrict__ A,
    const unsigned short* __restrict__ BT,
    unsigned short* __restrict__ outb,
    float* __restrict__ outf,
    const float* __restrict__ bias,
    int M, int N, int K, int mode, float scale)
{
    const int m0   = blockIdx.x * 32;
    const int n0   = blockIdx.y * 64;
    const int lane = threadIdx.x;
    const int lo   = lane & 15;
    const int hi   = lane >> 4;

    const v8f vzero = {0.f,0.f,0.f,0.f,0.f,0.f,0.f,0.f};
    v8f acc[2][4];
    #pragma unroll
    for (int mt = 0; mt < 2; ++mt)
        #pragma unroll
        for (int nt = 0; nt < 4; ++nt) acc[mt][nt] = vzero;

    const unsigned short* arow[2];
    const unsigned short* brow[4];
    #pragma unroll
    for (int mt = 0; mt < 2; ++mt) arow[mt] = A  + (size_t)(m0 + mt*16 + lo) * K;
    #pragma unroll
    for (int nt = 0; nt < 4; ++nt) brow[nt] = BT + (size_t)(n0 + nt*16 + lo) * K;

    for (int kk = 0; kk < K; kk += 32) {
        FragH fa[2], fb[4];
        #pragma unroll
        for (int mt = 0; mt < 2; ++mt) {
            // A frag: row = lo, k(e) = hi*8 + (e<8 ? e : e+8)
            const unsigned short* ap = arow[mt] + kk + hi * 8;
            fa[mt].q[0] = *(const uint4*)(ap);
            fa[mt].q[1] = *(const uint4*)(ap + 16);
        }
        #pragma unroll
        for (int nt = 0; nt < 4; ++nt) {
            // B frag: col = lo, k(e) = hi*8 + (e<8 ? e : e+8)
            const unsigned short* bp = brow[nt] + kk + hi * 8;
            fb[nt].q[0] = *(const uint4*)(bp);
            fb[nt].q[1] = *(const uint4*)(bp + 16);
        }
        #pragma unroll
        for (int mt = 0; mt < 2; ++mt)
            #pragma unroll
            for (int nt = 0; nt < 4; ++nt)
                acc[mt][nt] = __builtin_amdgcn_wmma_f32_16x16x32_bf16(
                    false, fa[mt].b, false, fb[nt].b, (short)0, acc[mt][nt], false, false);
    }

    #pragma unroll
    for (int mt = 0; mt < 2; ++mt) {
        #pragma unroll
        for (int nt = 0; nt < 4; ++nt) {
            int c = n0 + nt*16 + lo;
            if (mode == 3) {
                float bv = bias[c];
                #pragma unroll
                for (int r = 0; r < 8; ++r) {
                    int row = m0 + mt*16 + r + 8*hi;     // C layout: row = r + 8*(lane>=16)
                    outf[(size_t)row * N + c] = acc[mt][nt][r] + bv;
                }
            } else {
                #pragma unroll
                for (int r = 0; r < 8; ++r) {
                    int row = m0 + mt*16 + r + 8*hi;
                    float v = acc[mt][nt][r] * scale;
                    size_t idx;
                    if (mode == 2) idx = (size_t)c * M + row;                      // V^T [H*64][S]
                    else           idx = ((size_t)(c >> 6) * M + row) * 64 + (c & 63); // [H][S][64]
                    outb[idx] = f2bf(v);
                }
            }
        }
    }
}

// -------- causal flash attention: 4 waves/block share K/V chunks in LDS --------
// K/V double-buffered via the gfx1250 async-to-LDS engine so the DMA for chunk
// i+1 runs under the WMMAs of chunk i (4 async issues/wave/chunk; async loads
// complete in order, so s_wait_asynccnt 4 drains exactly the previous chunk).
__device__ __forceinline__ void stage_chunk_async(
    const unsigned short* Kb, const unsigned short* Vt,
    int h, int S, int kc, int tid,
    unsigned int ktb, unsigned int vtb)
{
    // K chunk: 32 keys x 64 dims bf16 = 4KB, contiguous in [H][S][64]
    unsigned long long kg =
        (unsigned long long)(uintptr_t)(Kb + ((size_t)h * S + kc) * 64) + (unsigned)(tid * 32);
    unsigned int kl = ktb + (unsigned)(tid * 32);
    asm volatile("global_load_async_to_lds_b128 %0, %1, off"
                 :: "v"(kl), "v"(kg) : "memory");
    asm volatile("global_load_async_to_lds_b128 %0, %1, off offset:16"
                 :: "v"(kl), "v"(kg) : "memory");
    // V chunk: 64 dim-rows x 64B, row stride S*2 in [H][64][S]
    #pragma unroll
    for (int r2 = 0; r2 < 2; ++r2) {
        int i   = tid + r2 * 128;
        int row = i >> 2;
        int o16 = (i & 3) * 16;
        unsigned long long vg =
            (unsigned long long)(uintptr_t)(Vt + ((size_t)(h * 64 + row)) * S + kc) + (unsigned)o16;
        unsigned int vl = vtb + (unsigned)(row * 64 + o16);
        asm volatile("global_load_async_to_lds_b128 %0, %1, off"
                     :: "v"(vl), "v"(vg) : "memory");
    }
}

__global__ __launch_bounds__(128) void attn_wmma(
    const unsigned short* __restrict__ Qb,   // [H][S][64] bf16, pre-scaled by 1/8
    const unsigned short* __restrict__ Kb,   // [H][S][64] bf16
    const unsigned short* __restrict__ Vt,   // [H][64][S] bf16
    unsigned short* __restrict__ ctx,        // [S][768] bf16
    int S)
{
    __shared__ __align__(16) unsigned short ktile[2][32 * 64];   // [key][dim]  2x4KB
    __shared__ __align__(16) unsigned short vtile[2][64 * 32];   // [dim][key]  2x4KB
    __shared__ __align__(16) unsigned short pl[4][16 * 32];      // per-wave P  4KB

    const int h    = blockIdx.y;
    const int tid  = threadIdx.x;
    const int wid  = tid >> 5;
    const int lane = tid & 31;
    const int lo   = lane & 15;
    const int hi   = lane >> 4;
    const int b8   = hi * 8;
    const int k16  = hi * 16;
    const int q0   = blockIdx.x * 64 + wid * 16;   // this wave's query tile

    // Q A-fragments for kdim [0,32) and [32,64)
    FragH qa0, qa1;
    const unsigned short* qp = Qb + ((size_t)h * S + q0 + lo) * 64;
    qa0.q[0] = *(const uint4*)(qp + b8);
    qa0.q[1] = *(const uint4*)(qp + b8 + 16);
    qa1.q[0] = *(const uint4*)(qp + 32 + b8);
    qa1.q[1] = *(const uint4*)(qp + 32 + b8 + 16);

    float rmax[8], rsum[8];
    v8f o[4];
    const v8f vzero = {0.f,0.f,0.f,0.f,0.f,0.f,0.f,0.f};
    #pragma unroll
    for (int r = 0; r < 8; ++r) { rmax[r] = -__builtin_inff(); rsum[r] = 0.f; }
    #pragma unroll
    for (int t = 0; t < 4; ++t) o[t] = vzero;

    const unsigned int ktb[2] = { (unsigned int)(uintptr_t)(void*)ktile[0],
                                  (unsigned int)(uintptr_t)(void*)ktile[1] };
    const unsigned int vtb[2] = { (unsigned int)(uintptr_t)(void*)vtile[0],
                                  (unsigned int)(uintptr_t)(void*)vtile[1] };
    unsigned short* plw = pl[wid];

    const int nchunks = (blockIdx.x * 64 + 64) / 32;   // block-wide causal key range
    stage_chunk_async(Kb, Vt, h, S, 0, tid, ktb[0], vtb[0]);   // prologue: chunk 0

    for (int i = 0; i < nchunks; ++i) {
        const int kc  = i * 32;
        const int cur = i & 1;
        const bool more = (i + 1 < nchunks);
        if (more) {
            // issue chunk i+1 into the other buffer; it flies under this chunk's math
            stage_chunk_async(Kb, Vt, h, S, kc + 32, tid, ktb[cur ^ 1], vtb[cur ^ 1]);
            asm volatile("s_wait_asynccnt 0x4" ::: "memory");  // drain chunk i only
        } else {
            asm volatile("s_wait_asynccnt 0x0" ::: "memory");
        }
        __syncthreads();                               // chunk i visible block-wide

        if (kc < q0 + 16) {                            // wave-uniform causal pruning
            const unsigned short* kt = ktile[cur];
            const unsigned short* vt = vtile[cur];
            // ---- scores: two 16x16 tiles over 32 keys, kdim=64 in 2 WMMAs each ----
            v8f s[2];
            #pragma unroll
            for (int t = 0; t < 2; ++t) {
                const unsigned short* kp = kt + (t * 16 + lo) * 64 + b8;
                FragH kb0, kb1;
                kb0.q[0] = *(const uint4*)(kp);
                kb0.q[1] = *(const uint4*)(kp + 16);
                kb1.q[0] = *(const uint4*)(kp + 32);
                kb1.q[1] = *(const uint4*)(kp + 48);
                v8f z = vzero;
                z    = __builtin_amdgcn_wmma_f32_16x16x32_bf16(false, qa0.b, false, kb0.b,
                                                               (short)0, z, false, false);
                s[t] = __builtin_amdgcn_wmma_f32_16x16x32_bf16(false, qa1.b, false, kb1.b,
                                                               (short)0, z, false, false);
            }
            // ---- causal mask (diagonal-touching chunk only) ----
            if (kc + 31 > q0) {
                #pragma unroll
                for (int t = 0; t < 2; ++t) {
                    int colg = kc + t * 16 + lo;
                    #pragma unroll
                    for (int r = 0; r < 8; ++r) {
                        int rowg = q0 + r + 8 * hi;
                        if (colg > rowg) s[t][r] = -__builtin_inff();
                    }
                }
            }
            // ---- online softmax (row stats via xor-shuffles in 16-lane halves) ----
            float p0[8], p1[8];
            #pragma unroll
            for (int r = 0; r < 8; ++r) {
                float cm = fmaxf(s[0][r], s[1][r]);
                cm = fmaxf(cm, __shfl_xor(cm, 1));
                cm = fmaxf(cm, __shfl_xor(cm, 2));
                cm = fmaxf(cm, __shfl_xor(cm, 4));
                cm = fmaxf(cm, __shfl_xor(cm, 8));
                float nm    = fmaxf(rmax[r], cm);
                float alpha = __expf(rmax[r] - nm);
                rmax[r] = nm;
                p0[r] = __expf(s[0][r] - nm);
                p1[r] = __expf(s[1][r] - nm);
                float cs = p0[r] + p1[r];
                cs += __shfl_xor(cs, 1);
                cs += __shfl_xor(cs, 2);
                cs += __shfl_xor(cs, 4);
                cs += __shfl_xor(cs, 8);
                rsum[r] = rsum[r] * alpha + cs;
                #pragma unroll
                for (int t = 0; t < 4; ++t) o[t][r] *= alpha;
            }
            // ---- P: C-layout regs -> LDS (16x32 bf16) -> A-fragment (per-wave) ----
            #pragma unroll
            for (int r = 0; r < 8; ++r) {
                int row = r + 8 * hi;
                plw[row * 32 + lo]      = f2bf(p0[r]);
                plw[row * 32 + 16 + lo] = f2bf(p1[r]);
            }
            asm volatile("s_wait_dscnt 0x0" ::: "memory");   // intra-wave RAW through LDS
            FragH pa;
            {
                const unsigned short* pp = plw + lo * 32 + b8;
                pa.q[0] = *(const uint4*)(pp);
                pa.q[1] = *(const uint4*)(pp + 16);
            }
            // ---- PV: accumulate 16x64 context in 4 tiles from shared V ----
            #pragma unroll
            for (int t = 0; t < 4; ++t) {
                const unsigned short* vp = vt + (t * 16 + lo) * 32 + b8;
                FragH vb;
                vb.q[0] = *(const uint4*)(vp);
                vb.q[1] = *(const uint4*)(vp + 16);
                o[t] = __builtin_amdgcn_wmma_f32_16x16x32_bf16(false, pa.b, false, vb.b,
                                                               (short)0, o[t], false, false);
            }
        }
        __syncthreads();   // all reads of buf[cur] done before it is re-staged (iter i+1)
    }

    // ---- normalize and emit ctx (bf16, [S][768]) ----
    #pragma unroll
    for (int r = 0; r < 8; ++r) {
        float inv = 1.f / rsum[r];
        int row = q0 + r + 8 * hi;
        #pragma unroll
        for (int t = 0; t < 4; ++t) {
            int col = h * 64 + t * 16 + lo;
            ctx[(size_t)row * 768 + col] = f2bf(o[t][r] * inv);
        }
    }
}

// ---------------- launch ----------------

extern "C" void kernel_launch(void* const* d_in, const int* in_sizes, int n_in,
                              void* d_out, int out_size, void* d_ws, size_t ws_size,
                              hipStream_t stream) {
    (void)in_sizes; (void)n_in; (void)out_size; (void)ws_size;
    const float* x  = (const float*)d_in[0];
    const float* wq = (const float*)d_in[1];
    const float* wk = (const float*)d_in[2];
    const float* wv = (const float*)d_in[3];
    const float* wo = (const float*)d_in[4];
    const float* bo = (const float*)d_in[5];
    float* out = (float*)d_out;

    const int S = 4096, D = 768, H = 12;
    char* ws = (char*)d_ws;
    const size_t xbytes = (size_t)S * D * 2;   // 6 MB per S x D bf16 tensor
    const size_t wbytes = (size_t)D * D * 2;   // 1.125 MB per weight
    unsigned short* xb  = (unsigned short*)ws; ws += xbytes;
    unsigned short* wqT = (unsigned short*)ws; ws += wbytes;
    unsigned short* wkT = (unsigned short*)ws; ws += wbytes;
    unsigned short* wvT = (unsigned short*)ws; ws += wbytes;
    unsigned short* woT = (unsigned short*)ws; ws += wbytes;
    unsigned short* Qb  = (unsigned short*)ws; ws += xbytes;
    unsigned short* Kb  = (unsigned short*)ws; ws += xbytes;
    unsigned short* Vt  = (unsigned short*)ws; ws += xbytes;
    unsigned short* ctx = (unsigned short*)ws; ws += xbytes;

    const int nx = S * D;
    cvt_f32_bf16<<<(nx + 255) / 256, 256, 0, stream>>>(x, xb, nx);
    const int nw = D * D;
    const int gw = (nw + 255) / 256;
    cvt_transpose_bf16<<<gw, 256, 0, stream>>>(wq, wqT, D, D);
    cvt_transpose_bf16<<<gw, 256, 0, stream>>>(wk, wkT, D, D);
    cvt_transpose_bf16<<<gw, 256, 0, stream>>>(wv, wvT, D, D);
    cvt_transpose_bf16<<<gw, 256, 0, stream>>>(wo, woT, D, D);

    dim3 gproj(S / 32, D / 64);
    // Q pre-scaled by 1/sqrt(HD)=0.125 (folds the softmax temperature into the GEMM)
    gemm_bf16_wmma<<<gproj, 32, 0, stream>>>(xb, wqT, Qb, nullptr, nullptr, S, D, D, 0, 0.125f);
    gemm_bf16_wmma<<<gproj, 32, 0, stream>>>(xb, wkT, Kb, nullptr, nullptr, S, D, D, 1, 1.0f);
    gemm_bf16_wmma<<<gproj, 32, 0, stream>>>(xb, wvT, Vt, nullptr, nullptr, S, D, D, 2, 1.0f);

    attn_wmma<<<dim3(S / 64, H), 128, 0, stream>>>(Qb, Kb, Vt, ctx, S);

    gemm_bf16_wmma<<<gproj, 32, 0, stream>>>(ctx, woT, nullptr, out, bo, S, D, D, 3, 1.0f);
}
